// FNO3D_RCLN_44341242363946
// MI455X (gfx1250) — matrix operand order repaired and checked by reading, not compiled, not run on hardware
//
#include <hip/hip_runtime.h>

#define S 262144        // 64^3
#define NROWS 1048576   // B * S
#define NUV 0.000185f
#define DXC 0.09817477042f   // 2*pi/64

typedef __attribute__((ext_vector_type(16))) unsigned short v16us;
typedef __attribute__((ext_vector_type(16))) __bf16 v16bf;
typedef __attribute__((ext_vector_type(8)))  float  v8f;
typedef __attribute__((ext_vector_type(4)))  unsigned int v4u;
typedef __attribute__((ext_vector_type(8)))  int v8i;
typedef __attribute__((ext_vector_type(4)))  int v4i;

static __device__ __forceinline__ unsigned short f2bf(float f) {
  union { float f; unsigned int u; } v; v.f = f;
  unsigned int u = v.u;
  unsigned int r = u + 0x7FFFu + ((u >> 16) & 1u);   // round-to-nearest-even
  return (unsigned short)(r >> 16);
}

static __device__ __forceinline__ float fgelu(float x) {
  float x3 = x * x * x;
  return 0.5f * x * (1.0f + tanhf(0.7978845608f * (x + 0.044715f * x3)));
}

// TDM: DMA a contiguous run of `nelem` f32 from global into LDS at byte
// offset `lds_off`.  D# per CDNA5 ISA 8.3/8.4: group0 = count|lds|global|type,
// group1 = data_size=4B, tensor_dim0=nelem, 1-row tile; groups 2/3 zero (2-D).
// This toolchain exposes the 6-arg builtin (g0, g1, g2, g3, g_extra, cpol).
static __device__ __forceinline__ void tdm_load_f32(const void* gaddr,
                                                    unsigned lds_off,
                                                    unsigned nelem) {
  unsigned long long ga = (unsigned long long)gaddr;
  v4u g0;
  g0[0] = 1u;                                             // count=1 (valid user D#)
  g0[1] = lds_off;                                        // lds_addr (bytes)
  g0[2] = (unsigned)(ga & 0xFFFFFFFFu);                   // global_addr[31:0]
  g0[3] = (unsigned)((ga >> 32) & 0x1FFFFFFu) | (2u << 30); // [56:32] | type=2
  v8i g1;
  g1[0] = (int)(2u << 16);                                // data_size=4B, no mask
  g1[1] = (int)((nelem & 0xFFFFu) << 16);                 // tensor_dim0[15:0]
  g1[2] = (int)((nelem >> 16) & 0xFFFFu) | (1 << 16);     // td0[31:16] | tensor_dim1=1
  g1[3] = (int)((nelem & 0xFFFFu) << 16);                 // td1 hi=0 | tile_dim0=nelem
  g1[4] = 1;                                              // tile_dim1=1, tile_dim2=0
  g1[5] = (int)nelem;                                     // tensor_dim0_stride[31:0]
  g1[6] = 0;
  g1[7] = 0;
  v4i z4 = {0, 0, 0, 0};
  v8i z8 = {0, 0, 0, 0, 0, 0, 0, 0};
  __builtin_amdgcn_tensor_load_to_lds(g0, g1, z4, z4, z8, 0);
}

// ---------------- fc0: (B,3,D,H,W) -> x (N,32) channels-last -----------------
__global__ void fno_fc0(const float* __restrict__ u, const float* __restrict__ w,
                        const float* __restrict__ b, float* __restrict__ x) {
  int tid = blockIdx.x * blockDim.x + threadIdx.x;   // N*32 threads
  int o = tid & 31;
  int n = tid >> 5;
  int bb = n >> 18;
  int s  = n & (S - 1);
  const float* ub = u + (size_t)bb * 3 * S + s;
  float acc = b[o];
  acc += ub[0]             * w[o];
  acc += ub[S]             * w[32 + o];
  acc += ub[2 * (size_t)S] * w[64 + o];
  x[tid] = acc;
}

// ---------------- forward truncated DFT along x (rfft, 8 modes) --------------
__global__ void fno_dftx(const float* __restrict__ x, float2* __restrict__ F1) {
  int tid = blockIdx.x * blockDim.x + threadIdx.x;   // 4*4096*8*32
  int c   = tid & 31;
  int kx  = (tid >> 5) & 7;
  int bzy = tid >> 8;
  const float* xp = x + (size_t)bzy * 64 * 32 + c;
  __builtin_prefetch(xp, 0, 1);                      // global_prefetch_b8
  float re = 0.f, im = 0.f;
  for (int i = 0; i < 64; ++i) {
    int t = (kx * i) & 63;
    float sn, cs; __sincosf((float)t * DXC, &sn, &cs);
    float v = xp[(size_t)i * 32];
    re += v * cs;
    im -= v * sn;                       // e^{-i theta}
  }
  F1[tid] = make_float2(re, im);
}

// ---------------- forward truncated DFT along y (16 slots) -------------------
__global__ void fno_dfty(const float2* __restrict__ F1, float2* __restrict__ F2) {
  int tid = blockIdx.x * blockDim.x + threadIdx.x;   // 4*64*16*8*32
  int c   = tid & 31;
  int kx  = (tid >> 5) & 7;
  int kyi = (tid >> 8) & 15;
  int bz  = tid >> 12;
  int ky  = kyi < 8 ? kyi : kyi + 48;
  const float2* p = F1 + (size_t)bz * 64 * 256 + kx * 32 + c;  // stride per y = 256
  float re = 0.f, im = 0.f;
  for (int y = 0; y < 64; ++y) {
    int t = (ky * y) & 63;
    float sn, cs; __sincosf((float)t * DXC, &sn, &cs);
    float2 f = p[(size_t)y * 256];
    re += f.x * cs + f.y * sn;
    im += f.y * cs - f.x * sn;
  }
  F2[tid] = make_float2(re, im);
}

// ---------------- forward truncated DFT along z (16 slots) -------------------
__global__ void fno_dftz(const float2* __restrict__ F2, float2* __restrict__ F3) {
  int tid = blockIdx.x * blockDim.x + threadIdx.x;   // 4*16*16*8*32
  int c   = tid & 31;
  int kx  = (tid >> 5) & 7;
  int kyi = (tid >> 8) & 15;
  int kzi = (tid >> 12) & 15;
  int b   = tid >> 16;
  int kz  = kzi < 8 ? kzi : kzi + 48;
  const float2* p = F2 + (size_t)b * 64 * 4096 + kyi * 256 + kx * 32 + c; // z stride 4096
  float re = 0.f, im = 0.f;
  for (int z = 0; z < 64; ++z) {
    int t = (kz * z) & 63;
    float sn, cs; __sincosf((float)t * DXC, &sn, &cs);
    float2 f = p[(size_t)z * 4096];
    re += f.x * cs + f.y * sn;
    im += f.y * cs - f.x * sn;
  }
  F3[tid] = make_float2(re, im);
}

// ---------------- per-mode 32x32 complex channel mix -------------------------
__global__ void fno_spec(const float2* __restrict__ F3, const float* __restrict__ w,
                         float2* __restrict__ F4) {
  int tid = blockIdx.x * blockDim.x + threadIdx.x;   // 4*16*16*8*32
  int o   = tid & 31;
  int kx  = (tid >> 5) & 7;
  int kyi = (tid >> 8) & 15;
  int kzi = (tid >> 12) & 15;
  int b   = tid >> 16;
  int corner = (kzi >= 8 ? 1 : 0) + (kyi >= 8 ? 2 : 0);
  int mm = (kzi & 7) * 64 + (kyi & 7) * 8 + kx;
  const float2* fp = F3 + ((size_t)(b * 16 + kzi) * 4096 + kyi * 256 + kx * 32);
  size_t wb = ((size_t)(corner * 1024 + o) * 512 + mm) * 2;
  float re = 0.f, im = 0.f;
  for (int c = 0; c < 32; ++c) {
    float2 f = fp[c];
    float wr = w[wb + (size_t)c * 32768];
    float wi = w[wb + (size_t)c * 32768 + 1];
    re += f.x * wr - f.y * wi;
    im += f.x * wi + f.y * wr;
  }
  F4[tid] = make_float2(re, im);
}

// ---------------- inverse DFT along z ----------------------------------------
__global__ void fno_idftz(const float2* __restrict__ F4, float2* __restrict__ G1) {
  int tid = blockIdx.x * blockDim.x + threadIdx.x;   // 4*64*16*8*32
  int o   = tid & 31;
  int kx  = (tid >> 5) & 7;
  int kyi = (tid >> 8) & 15;
  int z   = (tid >> 12) & 63;
  int b   = tid >> 18;
  const float2* p = F4 + (size_t)b * 16 * 4096 + kyi * 256 + kx * 32 + o; // kz stride 4096
  float re = 0.f, im = 0.f;
  #pragma unroll
  for (int kzi = 0; kzi < 16; ++kzi) {
    int kz = kzi < 8 ? kzi : kzi + 48;
    int t = (kz * z) & 63;
    float sn, cs; __sincosf((float)t * DXC, &sn, &cs);
    float2 f = p[(size_t)kzi * 4096];
    re += f.x * cs - f.y * sn;          // e^{+i theta}
    im += f.x * sn + f.y * cs;
  }
  G1[tid] = make_float2(re, im);
}

// ---------------- inverse DFT along y ----------------------------------------
__global__ void fno_idfty(const float2* __restrict__ G1, float2* __restrict__ G2) {
  int tid = blockIdx.x * blockDim.x + threadIdx.x;   // 4*64*64*8*32
  int o   = tid & 31;
  int kx  = (tid >> 5) & 7;
  int y   = (tid >> 8) & 63;
  int bz  = tid >> 14;
  const float2* p = G1 + (size_t)bz * 4096 + kx * 32 + o;   // ky stride 256
  float re = 0.f, im = 0.f;
  #pragma unroll
  for (int kyi = 0; kyi < 16; ++kyi) {
    int ky = kyi < 8 ? kyi : kyi + 48;
    int t = (ky * y) & 63;
    float sn, cs; __sincosf((float)t * DXC, &sn, &cs);
    float2 f = p[(size_t)kyi * 256];
    re += f.x * cs - f.y * sn;
    im += f.x * sn + f.y * cs;
  }
  G2[tid] = make_float2(re, im);
}

// ---------------- inverse rfft along x, fused with +pointwise, gelu ----------
__global__ void fno_irfft(const float2* __restrict__ G2, const float* __restrict__ yb,
                          float* __restrict__ xout, int dogelu) {
  int tid = blockIdx.x * blockDim.x + threadIdx.x;   // N*32
  int o   = tid & 31;
  int xi  = (tid >> 5) & 63;
  int bzy = tid >> 11;
  const float2* p = G2 + (size_t)bzy * 256 + o;      // kx stride 32
  __builtin_prefetch(p, 0, 1);
  float acc = p[0].x;                                 // k = 0 (real weight 1)
  #pragma unroll
  for (int k = 1; k < 8; ++k) {
    int t = (k * xi) & 63;
    float sn, cs; __sincosf((float)t * DXC, &sn, &cs);
    float2 f = p[(size_t)k * 32];
    acc += 2.0f * (f.x * cs - f.y * sn);              // + hermitian conjugate
  }
  acc *= (1.0f / 262144.0f);                          // irfftn normalization
  size_t idx = ((size_t)bzy * 64 + xi) * 32 + o;
  float v = acc + yb[idx];
  xout[idx] = dogelu ? fgelu(v) : v;
}

// ---------------- pointwise 32->32 GEMM via bf16 WMMA + TDM weight stage -----
// dynamic LDS: [0,1024) floats = W staged by TDM
__global__ void __launch_bounds__(256)
fno_pw(const float* __restrict__ X, const float* __restrict__ W,
       const float* __restrict__ bias, float* __restrict__ Y) {
  extern __shared__ float smem[];
  if (threadIdx.x < 32) {                       // wave 0 issues the DMA
    tdm_load_f32(W, 0u, 1024u);                 // 32x32 f32 -> LDS[0]
    __builtin_amdgcn_s_wait_tensorcnt((short)0);
  }
  __syncthreads();

  int lane = threadIdx.x & 31;
  int wid  = blockIdx.x * (blockDim.x >> 5) + (threadIdx.x >> 5);
  size_t row0 = (size_t)wid * 16;
  int m  = lane & 15;
  int hi = (lane >> 4) & 1;
  int koff = hi ? 8 : 0;          // A: K-half per lane group
  int kb   = hi ? 16 : 0;         // B: K-half per lane group

  const float* xr = X + (row0 + m) * 32;
  v16us au;
  #pragma unroll
  for (int e = 0; e < 8; ++e)  au[e] = f2bf(xr[koff + e]);
  #pragma unroll
  for (int e = 8; e < 16; ++e) au[e] = f2bf(xr[koff + 8 + e]);   // K = koff+16+(e-8)
  v16bf a = __builtin_bit_cast(v16bf, au);

  v16us b0u, b1u;
  #pragma unroll
  for (int e = 0; e < 16; ++e) {
    int k = e < 8 ? koff + e : koff + 8 + e;
    b0u[e] = f2bf(smem[k * 32 + m]);
    b1u[e] = f2bf(smem[k * 32 + 16 + m]);
  }
  v16bf b0 = __builtin_bit_cast(v16bf, b0u);
  v16bf b1 = __builtin_bit_cast(v16bf, b1u);

  v8f c0 = {}; v8f c1 = {};
  c0 = __builtin_amdgcn_wmma_f32_16x16x32_bf16(false, a, false, b0, (short)0, c0, false, false);
  c1 = __builtin_amdgcn_wmma_f32_16x16x32_bf16(false, a, false, b1, (short)0, c1, false, false);

  int mr = hi ? 8 : 0;
  float bi0 = bias[m], bi1 = bias[16 + m];
  #pragma unroll
  for (int r = 0; r < 8; ++r) {
    size_t row = row0 + mr + r;
    Y[row * 32 + m]      = c0[r] + bi0;
    Y[row * 32 + 16 + m] = c1[r] + bi1;
  }
}

// ---------------- fused head: fc1 (32->128) + gelu + fc2 (128->6) ------------
// dynamic LDS floats: W1 [0,4096)  W2 [4096,4864)  hs [4864, 4864+4*16*132)
__global__ void __launch_bounds__(128)
fno_head(const float* __restrict__ X, const float* __restrict__ W1,
         const float* __restrict__ B1, const float* __restrict__ W2,
         const float* __restrict__ B2, float* __restrict__ tau) {
  extern __shared__ float smem[];
  float* w1s = smem;
  float* w2s = smem + 4096;
  float* hs  = smem + 4864;
  if (threadIdx.x < 32) {                       // wave 0 issues both DMAs
    tdm_load_f32(W1, 0u, 4096u);                // 32x128 f32
    tdm_load_f32(W2, 16384u, 768u);             // 128x6  f32
    __builtin_amdgcn_s_wait_tensorcnt((short)0);
  }
  __syncthreads();

  int lane = threadIdx.x & 31;
  int wloc = threadIdx.x >> 5;
  int wid  = blockIdx.x * 4 + wloc;
  size_t row0 = (size_t)wid * 16;
  int m  = lane & 15;
  int hi = (lane >> 4) & 1;
  int koff = hi ? 8 : 0;
  int kb   = hi ? 16 : 0;
  int mr   = hi ? 8 : 0;

  // A tile from activations
  const float* xr = X + (row0 + m) * 32;
  v16us au;
  #pragma unroll
  for (int e = 0; e < 8; ++e)  au[e] = f2bf(xr[koff + e]);
  #pragma unroll
  for (int e = 8; e < 16; ++e) au[e] = f2bf(xr[koff + 8 + e]);
  v16bf a = __builtin_bit_cast(v16bf, au);

  float* ls = hs + wloc * 16 * 132;
  // fc1: 8 column tiles of 16 -> gelu -> LDS (16x128, stride 132)
  #pragma unroll
  for (int t = 0; t < 8; ++t) {
    v16us bu;
    #pragma unroll
    for (int e = 0; e < 16; ++e) bu[e] = f2bf(w1s[(e < 8 ? koff + e : koff + 8 + e) * 128 + t * 16 + m]);
    v16bf bt = __builtin_bit_cast(v16bf, bu);
    v8f h = {};
    h = __builtin_amdgcn_wmma_f32_16x16x32_bf16(false, a, false, bt, (short)0, h, false, false);
    float bv = B1[t * 16 + m];
    #pragma unroll
    for (int r = 0; r < 8; ++r)
      ls[(mr + r) * 132 + t * 16 + m] = fgelu(h[r] + bv);
  }
  __syncthreads();

  // fc2: K = 128 in 4 steps of 32; single 16-col tile (only 6 valid)
  v8f d = {};
  #pragma unroll
  for (int sst = 0; sst < 4; ++sst) {
    v16us a2u;
    #pragma unroll
    for (int e = 0; e < 8; ++e)  a2u[e] = f2bf(ls[m * 132 + sst * 32 + koff + e]);
    #pragma unroll
    for (int e = 8; e < 16; ++e) a2u[e] = f2bf(ls[m * 132 + sst * 32 + koff + 8 + e]);
    v16bf a2 = __builtin_bit_cast(v16bf, a2u);
    v16us b2u;
    #pragma unroll
    for (int e = 0; e < 16; ++e) {
      int k = sst * 32 + (e < 8 ? koff + e : koff + 8 + e);
      b2u[e] = (m < 6) ? f2bf(w2s[k * 6 + m]) : (unsigned short)0;
    }
    v16bf b2 = __builtin_bit_cast(v16bf, b2u);
    d = __builtin_amdgcn_wmma_f32_16x16x32_bf16(false, a2, false, b2, (short)0, d, false, false);
  }
  if (m < 6) {
    float bo = B2[m];
    #pragma unroll
    for (int r = 0; r < 8; ++r) {
      size_t row = row0 + mr + r;
      int bb = (int)(row >> 18);
      size_t s = row & (S - 1);
      tau[((size_t)bb * 6 + m) * S + s] = d[r] + bo;   // channel-first (B,6,D,H,W)
    }
  }
}

// ---------------- NS hard core stencil + 0.001*tau ---------------------------
__global__ void fno_dudt(const float* __restrict__ u, const float* __restrict__ tau,
                         float* __restrict__ du) {
  int tid = blockIdx.x * blockDim.x + threadIdx.x;   // B*3*S
  int s  = tid & (S - 1);
  int bi = tid >> 18;
  int i = bi % 3, b = bi / 3;
  int x = s & 63, y = (s >> 6) & 63, z = s >> 12;
  const float* ub = u + (size_t)b * 3 * S;
  const float* ui = ub + (size_t)i * S;
  float uc = ui[s];
  const float inv2dx = 1.0f / (2.0f * DXC);
  const float invdx2 = 1.0f / (DXC * DXC);
  float conv = 0.f, diff = 0.f;
  { // j = 0 : z
    int sp = (((z + 1) & 63) << 12) | (y << 6) | x;
    int sm = (((z + 63) & 63) << 12) | (y << 6) | x;
    float up = ui[sp], um = ui[sm];
    conv -= ub[s] * (up - um) * inv2dx;
    diff += up - 2.f * uc + um;
  }
  { // j = 1 : y
    int sp = (z << 12) | (((y + 1) & 63) << 6) | x;
    int sm = (z << 12) | (((y + 63) & 63) << 6) | x;
    float up = ui[sp], um = ui[sm];
    conv -= ub[S + s] * (up - um) * inv2dx;
    diff += up - 2.f * uc + um;
  }
  { // j = 2 : x
    int sp = (z << 12) | (y << 6) | ((x + 1) & 63);
    int sm = (z << 12) | (y << 6) | ((x + 63) & 63);
    float up = ui[sp], um = ui[sm];
    conv -= ub[2 * (size_t)S + s] * (up - um) * inv2dx;
    diff += up - 2.f * uc + um;
  }
  du[tid] = conv + NUV * diff * invdx2 + 0.001f * tau[((size_t)b * 6 + i) * S + s];
}

extern "C" void kernel_launch(void* const* d_in, const int* in_sizes, int n_in,
                              void* d_out, int out_size, void* d_ws, size_t ws_size,
                              hipStream_t stream) {
  (void)in_sizes; (void)n_in; (void)out_size; (void)ws_size;
  const float* u     = (const float*)d_in[0];
  const float* fc0w  = (const float*)d_in[1];
  const float* fc0b  = (const float*)d_in[2];
  const float* specw = (const float*)d_in[3];
  const float* wpw   = (const float*)d_in[4];
  const float* bpw   = (const float*)d_in[5];
  const float* fc1w  = (const float*)d_in[6];
  const float* fc1b  = (const float*)d_in[7];
  const float* fc2w  = (const float*)d_in[8];
  const float* fc2b  = (const float*)d_in[9];
  float* out = (float*)d_out;
  float* tau = out + 3145728;          // du_dt (B*3*S) first, then tau (B*6*S)

  float* ws = (float*)d_ws;
  float* xA = ws;
  float* xB = xA + 33554432;
  float* yb = xB + 33554432;
  float2* F1 = (float2*)(yb + 33554432);  // 4,194,304 complex
  float2* F2 = F1 + 4194304;              // 1,048,576
  float2* F3 = F2 + 1048576;              //   262,144
  float2* F4 = F3 + 262144;               //   262,144
  float2* G1 = F4 + 262144;               // 1,048,576
  float2* G2 = G1 + 1048576;              // 4,194,304

  fno_fc0<<<131072, 256, 0, stream>>>(u, fc0w, fc0b, xA);
  for (int l = 0; l < 4; ++l) {
    fno_dftx <<<16384, 256, 0, stream>>>(xA, F1);
    fno_dfty <<< 4096, 256, 0, stream>>>(F1, F2);
    fno_dftz <<< 1024, 256, 0, stream>>>(F2, F3);
    fno_spec <<< 1024, 256, 0, stream>>>(F3, specw + (size_t)l * 4194304, F4);
    fno_idftz<<< 4096, 256, 0, stream>>>(F4, G1);
    fno_idfty<<<16384, 256, 0, stream>>>(G1, G2);
    fno_pw   <<< 8192, 256, 4096, stream>>>(xA, wpw + l * 1024, bpw + l * 32, yb);
    fno_irfft<<<131072, 256, 0, stream>>>(G2, yb, xB, (l < 3) ? 1 : 0);
    float* t = xA; xA = xB; xB = t;
  }
  fno_head<<<16384, 128, 53248, stream>>>(xA, fc1w, fc1b, fc2w, fc2b, tau);
  fno_dudt<<<12288, 256, 0, stream>>>(u, tau, out);
}
